// LuongAttention_21337397526644
// MI455X (gfx1250) — matrix operand order repaired and checked by reading, not compiled, not run on hardware
//
#include <hip/hip_runtime.h>

// ---------------------------------------------------------------------------
// LuongAttention on gfx1250 (MI455X), restructured as:
//   c[b,q] = rowmax_k(s) + log(rowsum_k exp(s - rowmax))          (pass1, WMMA)
//   w[b,k] = sum_q exp(s[q,k] - c[b,q])                           (pass2, WMMA)
//   out[b,d] = (1/S) * sum_k w[b,k] * V[b,k,d]                    (pass3)
// where s[q,k] = sum_d V[b,q,d] * V[b,k,d].
//
// Pipeline: register-staged double-buffered LDS tiles (1 barrier/tile,
// next tile's global loads overlap current tile's 16-WMMA compute),
// two independent WMMA accumulator chains for XDL ILP.
// ---------------------------------------------------------------------------

#define BATCH 8
#define SEQ   2048
#define DIM   512
#define QT    128            // rows owned per workgroup (8 waves x 16 rows)
#define LDSROW 520           // 512 + 8 bf16 pad -> row stride 1040B, avoids bank conflicts
#define NTILES (SEQ / 16)

typedef __attribute__((ext_vector_type(16))) __bf16 v16bf;
typedef __attribute__((ext_vector_type(8)))  __bf16 v8bf;
typedef __attribute__((ext_vector_type(8)))  float  v8f;
typedef __attribute__((ext_vector_type(4)))  float  v4f;

__device__ __forceinline__ v8bf cvt8pair(v4f a, v4f b) {
  v8bf r;
  r[0] = (__bf16)a[0]; r[1] = (__bf16)a[1]; r[2] = (__bf16)a[2]; r[3] = (__bf16)a[3];
  r[4] = (__bf16)b[0]; r[5] = (__bf16)b[1]; r[6] = (__bf16)b[2]; r[7] = (__bf16)b[3];
  return r;
}

__device__ __forceinline__ v8bf cvt8(const float* __restrict__ p) {
  v4f x0 = *(const v4f*)p;
  v4f x1 = *(const v4f*)(p + 4);
  return cvt8pair(x0, x1);
}

__device__ __forceinline__ v16bf cat16(v8bf lo, v8bf hi) {
  return __builtin_shufflevector(lo, hi, 0,1,2,3,4,5,6,7,8,9,10,11,12,13,14,15);
}

// Preload this wave's 16-row A tile (rows 'row0 + (lane&15)', all of DIM) as
// 16 bf16 WMMA A-fragments. A layout (16-bit, 16x32): lane L holds M=L%16;
// elements 0..7 -> kd = 8*(L/16)+e ; elements 8..15 -> kd = 16+8*(L/16)+(e-8).
__device__ __forceinline__ void load_a_frags(const float* __restrict__ Vrow,
                                             int half, v16bf a[16]) {
#pragma unroll
  for (int dblk = 0; dblk < 16; ++dblk) {
    const float* p0 = Vrow + dblk * 32 + 8 * half;
    a[dblk] = cat16(cvt8(p0), cvt8(p0 + 16));
  }
}

// ---- register-pipelined staging of a 16-row f32 tile as bf16 into LDS -----
// 256 threads: thread t -> row t/16, d-chunk (t%16)*32 (32 contiguous elems).
struct StageRegs { v4f x[8]; };

__device__ __forceinline__ void stage_issue(StageRegs& sr,
                                            const float* __restrict__ Vb,
                                            int rowBase, int tid) {
  int r  = tid >> 4;
  int dc = (tid & 15) * 32;
  const float* src = Vb + (size_t)(rowBase + r) * DIM + dc;
#pragma unroll
  for (int j = 0; j < 8; ++j) sr.x[j] = *(const v4f*)(src + 4 * j);
}

__device__ __forceinline__ void stage_commit(const StageRegs& sr,
                                             __bf16* __restrict__ lds, int tid) {
  int r  = tid >> 4;
  int dc = (tid & 15) * 32;
  __bf16* dst = lds + r * LDSROW + dc;
#pragma unroll
  for (int j = 0; j < 4; ++j) {
    v8bf t = cvt8pair(sr.x[2 * j], sr.x[2 * j + 1]);
    *(v8bf*)(dst + 8 * j) = t;
  }
}

// B fragment from LDS tile: B is 32x16 (kd x col). Lane holds col = lane&15,
// elements 0..7 -> kd = 8*(lane/16)+e ; elements 8..15 -> kd = 16+8*(lane/16)+(e-8).
__device__ __forceinline__ v16bf load_b_frag(const __bf16* __restrict__ lds,
                                             int ln, int half, int dblk) {
  const __bf16* bp = lds + ln * LDSROW + dblk * 32 + 8 * half;
  v8bf lo = *(const v8bf*)bp;
  v8bf hi = *(const v8bf*)(bp + 16);
  return cat16(lo, hi);
}

// 16 chained WMMAs over DIM, split into two independent accumulator chains.
__device__ __forceinline__ void score_tile(const v16bf a[16],
                                           const __bf16* __restrict__ buf,
                                           int ln, int half,
                                           v8f& c0, v8f& c1) {
  c0 = v8f{};
  c1 = v8f{};
#pragma unroll
  for (int dblk = 0; dblk < 16; dblk += 2) {
    v16bf b0 = load_b_frag(buf, ln, half, dblk);
    c0 = __builtin_amdgcn_wmma_f32_16x16x32_bf16(false, a[dblk], false, b0,
                                                 (short)0, c0, false, false);
    v16bf b1 = load_b_frag(buf, ln, half, dblk + 1);
    c1 = __builtin_amdgcn_wmma_f32_16x16x32_bf16(false, a[dblk + 1], false, b1,
                                                 (short)0, c1, false, false);
  }
}

// ------------------------------- pass 1 ------------------------------------
// Row softmax stats: c[b,q] = m + log(l).
__global__ void __launch_bounds__(256)
luong_pass1_rowstats(const float* __restrict__ V, float* __restrict__ cvec) {
  const int b  = blockIdx.x / (SEQ / QT);
  const int q0 = (blockIdx.x % (SEQ / QT)) * QT;
  const int tid  = threadIdx.x;
  const int wave = tid >> 5;
  const int lane = tid & 31;
  const int half = lane >> 4;
  const int ln   = lane & 15;
  const float* Vb = V + (size_t)b * SEQ * DIM;

  __shared__ __bf16 kb[2][16 * LDSROW];

  v16bf a[16];
  load_a_frags(Vb + (size_t)(q0 + wave * 16 + ln) * DIM, half, a);

  float rm[8], rs[8];
#pragma unroll
  for (int r = 0; r < 8; ++r) { rm[r] = -1.0e30f; rs[r] = 0.0f; }

  StageRegs sr;
  stage_issue(sr, Vb, 0, tid);

  for (int t = 0; t < NTILES; ++t) {
    __bf16* buf = kb[t & 1];
    stage_commit(sr, buf, tid);          // waits on tile-t global loads here
    __syncthreads();
    if (t + 1 < NTILES)                  // issue next tile's loads, no wait:
      stage_issue(sr, Vb, (t + 1) * 16, tid);   // latency hidden by compute

    v8f c0, c1;
    score_tile(a, buf, ln, half, c0, c1);

    // online (max, sum-exp) per lane over its k columns
#pragma unroll
    for (int r = 0; r < 8; ++r) {
      float s  = c0[r] + c1[r];
      float nm = fmaxf(rm[r], s);
      rs[r] = rs[r] * __expf(rm[r] - nm) + __expf(s - nm);
      rm[r] = nm;
    }
  }

  // combine across the 16 lanes of each half (rows r + 8*half)
#pragma unroll
  for (int r = 0; r < 8; ++r) {
    float M = rm[r], Ssum = rs[r];
#pragma unroll
    for (int off = 8; off >= 1; off >>= 1) {
      float oM = __shfl_xor(M, off, 16);
      float oS = __shfl_xor(Ssum, off, 16);
      float nm = fmaxf(M, oM);
      Ssum = Ssum * __expf(M - nm) + oS * __expf(oM - nm);
      M = nm;
    }
    if (ln == 0) {
      int row = q0 + wave * 16 + 8 * half + r;
      cvec[(size_t)b * SEQ + row] = M + __logf(Ssum);
    }
  }
}

// ------------------------------- pass 2 ------------------------------------
// Column weights: w[b,k] = sum_q exp(s[q,k] - c[b,q]).
__global__ void __launch_bounds__(256)
luong_pass2_colweights(const float* __restrict__ V,
                       const float* __restrict__ cvec,
                       float* __restrict__ wvec) {
  const int b  = blockIdx.x / (SEQ / QT);
  const int k0 = (blockIdx.x % (SEQ / QT)) * QT;
  const int tid  = threadIdx.x;
  const int wave = tid >> 5;
  const int lane = tid & 31;
  const int half = lane >> 4;
  const int ln   = lane & 15;
  const float* Vb = V + (size_t)b * SEQ * DIM;

  __shared__ __bf16 qb[2][16 * LDSROW];

  v16bf a[16];                              // A = this wave's 16 key rows
  load_a_frags(Vb + (size_t)(k0 + wave * 16 + ln) * DIM, half, a);

  float wsum[8];
#pragma unroll
  for (int r = 0; r < 8; ++r) wsum[r] = 0.0f;

  StageRegs sr;
  stage_issue(sr, Vb, 0, tid);

  for (int t = 0; t < NTILES; ++t) {
    __bf16* buf = qb[t & 1];
    stage_commit(sr, buf, tid);
    __syncthreads();
    if (t + 1 < NTILES)
      stage_issue(sr, Vb, (t + 1) * 16, tid);

    float cq = cvec[(size_t)b * SEQ + t * 16 + ln];  // c for this lane's q col

    v8f c0, c1;
    score_tile(a, buf, ln, half, c0, c1);

#pragma unroll
    for (int r = 0; r < 8; ++r)
      wsum[r] += __expf((c0[r] + c1[r]) - cq);
  }

#pragma unroll
  for (int r = 0; r < 8; ++r) {
    float s = wsum[r];
#pragma unroll
    for (int off = 8; off >= 1; off >>= 1)
      s += __shfl_xor(s, off, 16);
    if (ln == 0) {
      int row = k0 + wave * 16 + 8 * half + r;
      wvec[(size_t)b * SEQ + row] = s;
    }
  }
}

// ------------------------------- pass 3 ------------------------------------
// out[b,d] = (1/S) * sum_k w[b,k] * V[b,k,d]
__global__ void __launch_bounds__(256)
luong_pass3_gemv(const float* __restrict__ V, const float* __restrict__ wvec,
                 float* __restrict__ out) {
  const int b = blockIdx.y;
  const int d = blockIdx.x * blockDim.x + threadIdx.x;
  const float* Vb = V + (size_t)b * SEQ * DIM;
  const float* wb = wvec + (size_t)b * SEQ;
  float acc = 0.0f;
  for (int k = 0; k < SEQ; ++k)
    acc += wb[k] * Vb[(size_t)k * DIM + d];
  out[(size_t)b * DIM + d] = acc * (1.0f / (float)SEQ);
}

// ---------------------------------------------------------------------------
extern "C" void kernel_launch(void* const* d_in, const int* in_sizes, int n_in,
                              void* d_out, int out_size, void* d_ws, size_t ws_size,
                              hipStream_t stream) {
  (void)in_sizes; (void)n_in; (void)out_size; (void)ws_size;
  const float* V = (const float*)d_in[0];
  float* out  = (float*)d_out;
  float* cvec = (float*)d_ws;                  // [BATCH][SEQ] f32
  float* wvec = cvec + (size_t)BATCH * SEQ;    // [BATCH][SEQ] f32  (128 KB total)

  dim3 blk(256);
  dim3 grid12(BATCH * (SEQ / QT));             // 8 * 16 = 128 workgroups

  luong_pass1_rowstats<<<grid12, blk, 0, stream>>>(V, cvec);
  luong_pass2_colweights<<<grid12, blk, 0, stream>>>(V, cvec, wvec);
  luong_pass3_gemv<<<dim3(DIM / 256, BATCH), blk, 0, stream>>>(V, wvec, out);
}
